// GCRNCell_26671746908718
// MI455X (gfx1250) — matrix operand order repaired and checked by reading, not compiled, not run on hardware
//
#include <hip/hip_runtime.h>

// ---------------------------------------------------------------------------
// GCRN cell for MI455X (gfx1250, wave32, WMMA).
// B=32, N=2048, D=64, H=4, C=16.
// Heavy matmuls: v_wmma_f32_16x16x32_bf16 (f32 accumulate), double-buffered
// LDS with register staging, one barrier per K-step.
// ---------------------------------------------------------------------------

typedef __attribute__((ext_vector_type(16))) __bf16 v16bf;
typedef __attribute__((ext_vector_type(8)))  float  v8f;
typedef unsigned int   u32;
typedef unsigned short u16;

static constexpr int Nn = 2048;
static constexpr int Dd = 64;
static constexpr int Bb = 32;

__device__ __forceinline__ u16 f2bf(float f) {
  u32 u = __float_as_uint(f);
  u32 r = (u + 0x7FFFu + ((u >> 16) & 1u)) >> 16;   // round-to-nearest-even
  return (u16)r;
}

// ---------------------------------------------------------------------------
// Kernel 1: adaptive adjacency  A_adp[h,w,:] = softmax(relu(E1[w,h]·E2[v,h])/4)
// ---------------------------------------------------------------------------
__launch_bounds__(256)
__global__ void adj_softmax_kernel(const float* __restrict__ E1,
                                   const float* __restrict__ E2,
                                   u16* __restrict__ Aadp) {
  __shared__ float e1s[16];
  __shared__ float red[256];
  const int h = blockIdx.x >> 11;
  const int w = blockIdx.x & (Nn - 1);
  const int tid = threadIdx.x;
  if (tid < 16) e1s[tid] = E1[((size_t)w * 4 + h) * 16 + tid];
  __syncthreads();

  float sc[8];
  float vmax = 0.f;                      // relu -> scores >= 0
#pragma unroll
  for (int j = 0; j < 8; ++j) {
    int v = tid * 8 + j;
    const float* e2 = E2 + ((size_t)v * 4 + h) * 16;
    float d = 0.f;
#pragma unroll
    for (int c = 0; c < 16; ++c) d += e1s[c] * e2[c];
    float s = fmaxf(d, 0.f) * 0.25f;     // /sqrt(16)
    sc[j] = s;
    vmax = fmaxf(vmax, s);
  }
  red[tid] = vmax; __syncthreads();
  for (int off = 128; off > 0; off >>= 1) {
    if (tid < off) red[tid] = fmaxf(red[tid], red[tid + off]);
    __syncthreads();
  }
  vmax = red[0]; __syncthreads();

  float sum = 0.f;
#pragma unroll
  for (int j = 0; j < 8; ++j) { sc[j] = __expf(sc[j] - vmax); sum += sc[j]; }
  red[tid] = sum; __syncthreads();
  for (int off = 128; off > 0; off >>= 1) {
    if (tid < off) red[tid] += red[tid + off];
    __syncthreads();
  }
  float inv = 1.f / red[0];

  uint4 out;
  u32* op = (u32*)&out;
#pragma unroll
  for (int p = 0; p < 4; ++p)
    op[p] = (u32)f2bf(sc[2 * p] * inv) | ((u32)f2bf(sc[2 * p + 1] * inv) << 16);
  *(uint4*)(Aadp + ((size_t)h * Nn + w) * Nn + tid * 8) = out;
}

// ---------------------------------------------------------------------------
// Kernel 2: propagation GEMM   out[b,w,c] = sum_v A[w,v] * X[b,v,c]
// A: [2048,2048] bf16.  X: [B,2048,64] bf16 (XF32=0) or f32 (XF32=1,
// converted to bf16 during LDS staging).  Output tile 128x64 per block
// (8 waves), K-step 32, double-buffered LDS, register-staged tiles.
// Epilogue (compile-time): optional f32 H-out + fused S (+)= coef * out.
// ---------------------------------------------------------------------------
template <int XF32, int INIT, int WRITE_H>
__launch_bounds__(256)
__global__ void prop_gemm_kernel(const u16* __restrict__ A,
                                 const void* __restrict__ Xv,
                                 float* __restrict__ Hout,
                                 float* __restrict__ S,
                                 const float* wpre, const float* wadp,
                                 int sel) {
  __shared__ u16 At[2][128][40];   // A tile 128x32 (+pad)
  __shared__ u16 Bt[2][64][40];    // X tile transposed: Bt[n][k] (+pad)

  const int tid  = threadIdx.x;
  const int lane = tid & 31;
  const int wv   = tid >> 5;
  const int b    = blockIdx.y;
  const int row0 = blockIdx.x * 128;
  const float coef = sel ? (0.25f * wadp[0]) : wpre[0];

  // staging geometry (per thread)
  const int ar0 = tid >> 2,  aq0 = tid & 3;         // A: uint4 (tid) and (tid+256)
  const int ar1 = (tid + 256) >> 2, aq1 = (tid + 256) & 3;
  const int kb4 = (tid >> 5) * 4;                   // B: 4(K) x 2(N) block
  const int nb2 = (tid & 31) * 2;

  const u16*   X16 = (const u16*)Xv;
  const float* X32 = (const float*)Xv;
  const size_t xbase = (size_t)b * (Nn * Dd);

  v8f acc[4];
#pragma unroll
  for (int nt = 0; nt < 4; ++nt)
#pragma unroll
    for (int i = 0; i < 8; ++i) acc[nt][i] = 0.f;

  uint4 ra0, ra1;          // staged A regs
  u32  bl0, bl1, bh0, bh1; // staged B regs (packed bf16 pairs)

  auto load_tile = [&](int k) {
    ra0 = *(const uint4*)(A + (size_t)(row0 + ar0) * Nn + k + aq0 * 8);
    ra1 = *(const uint4*)(A + (size_t)(row0 + ar1) * Nn + k + aq1 * 8);
    if constexpr (XF32) {
      const float* src = X32 + xbase + (size_t)k * Dd;
      float2 f0 = *(const float2*)(src + (kb4 + 0) * Dd + nb2);
      float2 f1 = *(const float2*)(src + (kb4 + 1) * Dd + nb2);
      float2 f2 = *(const float2*)(src + (kb4 + 2) * Dd + nb2);
      float2 f3 = *(const float2*)(src + (kb4 + 3) * Dd + nb2);
      bl0 = (u32)f2bf(f0.x) | ((u32)f2bf(f1.x) << 16);
      bl1 = (u32)f2bf(f2.x) | ((u32)f2bf(f3.x) << 16);
      bh0 = (u32)f2bf(f0.y) | ((u32)f2bf(f1.y) << 16);
      bh1 = (u32)f2bf(f2.y) | ((u32)f2bf(f3.y) << 16);
    } else {
      const u16* src = X16 + xbase + (size_t)k * Dd;
      u32 d0 = *(const u32*)(src + (kb4 + 0) * Dd + nb2);
      u32 d1 = *(const u32*)(src + (kb4 + 1) * Dd + nb2);
      u32 d2 = *(const u32*)(src + (kb4 + 2) * Dd + nb2);
      u32 d3 = *(const u32*)(src + (kb4 + 3) * Dd + nb2);
      bl0 = (d0 & 0xFFFFu) | (d1 << 16);
      bl1 = (d2 & 0xFFFFu) | (d3 << 16);
      bh0 = (d0 >> 16) | (d1 & 0xFFFF0000u);
      bh1 = (d2 >> 16) | (d3 & 0xFFFF0000u);
    }
  };
  auto store_tile = [&](int buf) {
    *(uint4*)&At[buf][ar0][aq0 * 8] = ra0;
    *(uint4*)&At[buf][ar1][aq1 * 8] = ra1;
    *(uint2*)&Bt[buf][nb2][kb4]     = make_uint2(bl0, bl1);
    *(uint2*)&Bt[buf][nb2 + 1][kb4] = make_uint2(bh0, bh1);
  };

  // fragment addresses (fixed per lane)
  const int fm  = wv * 16 + (lane & 15);
  const int fka = (lane >> 4) * 8;
  const int fnl = lane & 15;
  const int fkb = (lane >> 4) * 8;

  // prologue: tile 0
  load_tile(0);
  store_tile(0);
  __syncthreads();

  for (int t = 0; t < Nn / 32; ++t) {
    const int cur = t & 1;
    const bool has_next = (t + 1) < (Nn / 32);
    if (has_next) load_tile((t + 1) * 32);
    if (t + 2 < Nn / 32)   // keep A tiles hot in L2/L0 (reused by 32 batches)
      __builtin_prefetch(A + (size_t)(row0 + ar0) * Nn + (t + 2) * 32, 0, 1);

    // A fragment (16x32 bf16, ISA per-lane layout)
    v16bf af;
    {
      uint4 lo = *(const uint4*)&At[cur][fm][fka];       // K = kb..kb+7
      uint4 hi = *(const uint4*)&At[cur][fm][16 + fka];  // K = 16+kb..+7
      u32* au = reinterpret_cast<u32*>(&af);
      au[0] = lo.x; au[1] = lo.y; au[2] = lo.z; au[3] = lo.w;
      au[4] = hi.x; au[5] = hi.y; au[6] = hi.z; au[7] = hi.w;
    }
#pragma unroll
    for (int nt = 0; nt < 4; ++nt) {
      v16bf bf;
      uint4 lo = *(const uint4*)&Bt[cur][nt * 16 + fnl][fkb];
      uint4 hi = *(const uint4*)&Bt[cur][nt * 16 + fnl][fkb + 16];
      u32* bu = reinterpret_cast<u32*>(&bf);
      bu[0] = lo.x; bu[1] = lo.y; bu[2] = lo.z; bu[3] = lo.w;
      bu[4] = hi.x; bu[5] = hi.y; bu[6] = hi.z; bu[7] = hi.w;
      acc[nt] = __builtin_amdgcn_wmma_f32_16x16x32_bf16(
          false, af, false, bf, (short)0, acc[nt], false, false);
    }
    if (has_next) store_tile(cur ^ 1);
    __syncthreads();
  }

  // epilogue: C layout lanes0-15 M=i / lanes16-31 M=8+i, N=lane&15
  const int mh = (lane >> 4) * 8;
#pragma unroll
  for (int nt = 0; nt < 4; ++nt) {
#pragma unroll
    for (int i = 0; i < 8; ++i) {
      int m = row0 + wv * 16 + mh + i;
      int c = nt * 16 + fnl;
      size_t idx = ((size_t)b * Nn + m) * Dd + c;
      float v = acc[nt][i];
      if constexpr (WRITE_H) Hout[idx] = v;
      if constexpr (INIT) S[idx] = coef * v;
      else                S[idx] += coef * v;
    }
  }
}

// ---------------------------------------------------------------------------
// Kernel 3: gate linear  G = act( [x | S1 | S2] @ Wbf + alpha*bias )
// M = B*N = 65536 rows, N = 64, K = 192 (x bf16; S1,S2 f32->bf16 on the fly).
// Wbf rows 0..63 pre-scaled by alpha.  ACT: 0=sigmoid, 1=tanh.
// ---------------------------------------------------------------------------
template <int ACT>
__launch_bounds__(256)
__global__ void gate_linear_kernel(const u16* __restrict__ Xbf,
                                   const float* __restrict__ S1,
                                   const float* __restrict__ S2,
                                   const u16* __restrict__ Wbf,
                                   const float* __restrict__ bias,
                                   const float* wpre, const float* wadp,
                                   float* __restrict__ G) {
  __shared__ u16 At[128][40];
  __shared__ u16 Bt[64][40];
  const int tid  = threadIdx.x;
  const int lane = tid & 31;
  const int wv   = tid >> 5;
  const int row0 = blockIdx.x * 128;
  const float alpha = 2.0f * wpre[0] + wadp[0];

  const int kb4 = (tid >> 5) * 4;
  const int nb2 = (tid & 31) * 2;

  v8f acc[4];
#pragma unroll
  for (int nt = 0; nt < 4; ++nt)
#pragma unroll
    for (int i = 0; i < 8; ++i) acc[nt][i] = 0.f;

  for (int kb2 = 0; kb2 < 6; ++kb2) {
    const int seg  = kb2 >> 1;           // 0:x  1:S1  2:S2
    const int koff = (kb2 & 1) * 32;
    if (seg == 0) {
#pragma unroll
      for (int it = 0; it < 2; ++it) {
        int u = tid + it * 256;
        int r = u >> 2, q = u & 3;
        uint4 va = *(const uint4*)(Xbf + (size_t)(row0 + r) * Dd + koff + q * 8);
        *(uint4*)&At[r][q * 8] = va;
      }
    } else {
      const float* Sp = (seg == 1) ? S1 : S2;
#pragma unroll
      for (int it = 0; it < 4; ++it) {
        int u = tid + it * 256;          // float4 index over 128x8
        int r = u >> 3, q = u & 7;
        float4 f = *(const float4*)(Sp + (size_t)(row0 + r) * Dd + koff + q * 4);
        u32 p0 = (u32)f2bf(f.x) | ((u32)f2bf(f.y) << 16);
        u32 p1 = (u32)f2bf(f.z) | ((u32)f2bf(f.w) << 16);
        *(uint2*)&At[r][q * 4] = make_uint2(p0, p1);
      }
    }
    // stage W rows [kb2*32, +32) transposed (4x2 blocks, packed b64 stores)
    {
      const u16* src = Wbf + (size_t)kb2 * 32 * Dd;
      u32 d0 = *(const u32*)(src + (kb4 + 0) * Dd + nb2);
      u32 d1 = *(const u32*)(src + (kb4 + 1) * Dd + nb2);
      u32 d2 = *(const u32*)(src + (kb4 + 2) * Dd + nb2);
      u32 d3 = *(const u32*)(src + (kb4 + 3) * Dd + nb2);
      *(uint2*)&Bt[nb2][kb4] =
          make_uint2((d0 & 0xFFFFu) | (d1 << 16), (d2 & 0xFFFFu) | (d3 << 16));
      *(uint2*)&Bt[nb2 + 1][kb4] =
          make_uint2((d0 >> 16) | (d1 & 0xFFFF0000u), (d2 >> 16) | (d3 & 0xFFFF0000u));
    }
    __syncthreads();

    v16bf af;
    {
      int m  = wv * 16 + (lane & 15);
      int kb = (lane >> 4) * 8;
      uint4 lo = *(const uint4*)&At[m][kb];
      uint4 hi = *(const uint4*)&At[m][16 + kb];
      u32* au = reinterpret_cast<u32*>(&af);
      au[0] = lo.x; au[1] = lo.y; au[2] = lo.z; au[3] = lo.w;
      au[4] = hi.x; au[5] = hi.y; au[6] = hi.z; au[7] = hi.w;
    }
#pragma unroll
    for (int nt = 0; nt < 4; ++nt) {
      v16bf bf;
      int n  = nt * 16 + (lane & 15);
      int kb = (lane >> 4) * 8;
      uint4 lo = *(const uint4*)&Bt[n][kb];
      uint4 hi = *(const uint4*)&Bt[n][kb + 16];
      u32* bu = reinterpret_cast<u32*>(&bf);
      bu[0] = lo.x; bu[1] = lo.y; bu[2] = lo.z; bu[3] = lo.w;
      bu[4] = hi.x; bu[5] = hi.y; bu[6] = hi.z; bu[7] = hi.w;
      acc[nt] = __builtin_amdgcn_wmma_f32_16x16x32_bf16(
          false, af, false, bf, (short)0, acc[nt], false, false);
    }
    __syncthreads();
  }

  const int mh = (lane >> 4) * 8;
  const int nl = lane & 15;
#pragma unroll
  for (int nt = 0; nt < 4; ++nt) {
#pragma unroll
    for (int i = 0; i < 8; ++i) {
      int m = row0 + wv * 16 + mh + i;
      int c = nt * 16 + nl;
      float t = acc[nt][i] + alpha * bias[c];
      if constexpr (ACT) t = tanhf(t);
      else               t = 1.f / (1.f + __expf(-t));
      G[(size_t)m * Dd + c] = t;
    }
  }
}

// ---------------------------------------------------------------------------
// Elementwise helpers
// ---------------------------------------------------------------------------
__global__ void cvt_bf16_pk_kernel(const float* __restrict__ src,
                                   u16* __restrict__ dst, int n2) {
  int i = blockIdx.x * blockDim.x + threadIdx.x;
  if (i < n2) {
    float2 f = ((const float2*)src)[i];
    ((u32*)dst)[i] = (u32)f2bf(f.x) | ((u32)f2bf(f.y) << 16);
  }
}

__global__ void cvt_w_kernel(const float* __restrict__ W, u16* __restrict__ Wbf,
                             const float* wpre, const float* wadp) {
  int i = blockIdx.x * blockDim.x + threadIdx.x;   // over 192*64
  if (i < 192 * 64) {
    float alpha = 2.f * wpre[0] + wadp[0];
    float s = (i < 64 * 64) ? alpha : 1.f;         // fold alpha into W0 rows
    Wbf[i] = f2bf(W[i] * s);
  }
}

__global__ void rx_kernel(const float* __restrict__ r,
                          const float* __restrict__ h,
                          u16* __restrict__ rxbf, int n2) {
  int i = blockIdx.x * blockDim.x + threadIdx.x;
  if (i < n2) {
    float2 rr = ((const float2*)r)[i];
    float2 hh = ((const float2*)h)[i];
    ((u32*)rxbf)[i] = (u32)f2bf(rr.x * hh.x) | ((u32)f2bf(rr.y * hh.y) << 16);
  }
}

__global__ void final_kernel(const float* __restrict__ z,
                             const float* __restrict__ h,
                             const float* __restrict__ c,
                             float* __restrict__ out, int n) {
  int i = blockIdx.x * blockDim.x + threadIdx.x;
  if (i < n) out[i] = (1.f - z[i]) * h[i] + z[i] * c[i];
}

// ---------------------------------------------------------------------------
extern "C" void kernel_launch(void* const* d_in, const int* in_sizes, int n_in,
                              void* d_out, int out_size, void* d_ws, size_t ws_size,
                              hipStream_t stream) {
  (void)in_sizes; (void)n_in; (void)out_size; (void)ws_size;
  const float* hidden = (const float*)d_in[0];
  const float* A1f    = (const float*)d_in[1];
  const float* A2f    = (const float*)d_in[2];
  const float* E1     = (const float*)d_in[3];
  const float* E2     = (const float*)d_in[4];
  const float* Wz     = (const float*)d_in[5];
  const float* bz     = (const float*)d_in[6];
  const float* Wr     = (const float*)d_in[7];
  const float* br     = (const float*)d_in[8];
  const float* Wc     = (const float*)d_in[9];
  const float* bc     = (const float*)d_in[10];
  const float* wpre   = (const float*)d_in[11];
  const float* wadp   = (const float*)d_in[12];
  float* out = (float*)d_out;

  const size_t NN  = (size_t)Nn * Nn;          // 4.19M
  const size_t BND = (size_t)Bb * Nn * Dd;     // 4.19M

  char* ws = (char*)d_ws;
  size_t off = 0;
  auto alloc = [&](size_t bytes) -> char* {
    char* p = ws + off;
    off += (bytes + 255) & ~(size_t)255;
    return p;
  };
  u16*   A1bf = (u16*)alloc(NN * 2);
  u16*   A2bf = (u16*)alloc(NN * 2);
  u16*   Adp  = (u16*)alloc(4 * NN * 2);
  u16*   Xh   = (u16*)alloc(BND * 2);
  u16*   Rx   = (u16*)alloc(BND * 2);
  float* H1   = (float*)alloc(BND * 4);        // hop-1 result, kept in f32
  float* S1   = (float*)alloc(BND * 4);
  float* S2   = (float*)alloc(BND * 4);
  float* Zg   = (float*)alloc(BND * 4);
  float* Rg   = (float*)alloc(BND * 4);
  u16*   Wzbf = (u16*)alloc(192 * 64 * 2);
  u16*   Wrbf = (u16*)alloc(192 * 64 * 2);
  u16*   Wcbf = (u16*)alloc(192 * 64 * 2);
  float* Cg   = Rg;   // r gate dead after rx_kernel; reuse for c gate

  // --- one-time conversions + adaptive adjacency ---
  cvt_bf16_pk_kernel<<<(int)((NN / 2 + 255) / 256), 256, 0, stream>>>(A1f, A1bf, (int)(NN / 2));
  cvt_bf16_pk_kernel<<<(int)((NN / 2 + 255) / 256), 256, 0, stream>>>(A2f, A2bf, (int)(NN / 2));
  cvt_bf16_pk_kernel<<<(int)((BND / 2 + 255) / 256), 256, 0, stream>>>(hidden, Xh, (int)(BND / 2));
  cvt_w_kernel<<<(192 * 64 + 255) / 256, 256, 0, stream>>>(Wz, Wzbf, wpre, wadp);
  cvt_w_kernel<<<(192 * 64 + 255) / 256, 256, 0, stream>>>(Wr, Wrbf, wpre, wadp);
  cvt_w_kernel<<<(192 * 64 + 255) / 256, 256, 0, stream>>>(Wc, Wcbf, wpre, wadp);
  adj_softmax_kernel<<<4 * Nn, 256, 0, stream>>>(E1, E2, Adp);

  const u16* adjs[6] = {A1bf, A2bf, Adp, Adp + NN, Adp + 2 * NN, Adp + 3 * NN};
  const int  sels[6] = {0, 0, 1, 1, 1, 1};

  dim3 pgrid(Nn / 128, Bb);
  auto run_phase = [&](const u16* Xin) {
    for (int a = 0; a < 6; ++a) {
      // hop 1: H1 = A @ Xin (f32) ; S1 (+)= coef * H1
      if (a == 0)
        prop_gemm_kernel<0, 1, 1><<<pgrid, 256, 0, stream>>>(
            adjs[a], (const void*)Xin, H1, S1, wpre, wadp, sels[a]);
      else
        prop_gemm_kernel<0, 0, 1><<<pgrid, 256, 0, stream>>>(
            adjs[a], (const void*)Xin, H1, S1, wpre, wadp, sels[a]);
      // hop 2: S2 (+)= coef * (A @ H1)   (H1 f32 -> bf16 during staging)
      if (a == 0)
        prop_gemm_kernel<1, 1, 0><<<pgrid, 256, 0, stream>>>(
            adjs[a], (const void*)H1, nullptr, S2, wpre, wadp, sels[a]);
      else
        prop_gemm_kernel<1, 0, 0><<<pgrid, 256, 0, stream>>>(
            adjs[a], (const void*)H1, nullptr, S2, wpre, wadp, sels[a]);
    }
  };

  // --- phase 1: x = hidden -> z, r ---
  run_phase(Xh);
  gate_linear_kernel<0><<<(Bb * Nn) / 128, 256, 0, stream>>>(Xh, S1, S2, Wzbf, bz,
                                                             wpre, wadp, Zg);
  gate_linear_kernel<0><<<(Bb * Nn) / 128, 256, 0, stream>>>(Xh, S1, S2, Wrbf, br,
                                                             wpre, wadp, Rg);
  // --- phase 2: x = r * hidden -> c ---
  rx_kernel<<<(int)((BND / 2 + 255) / 256), 256, 0, stream>>>(Rg, hidden, Rx, (int)(BND / 2));
  run_phase(Rx);
  gate_linear_kernel<1><<<(Bb * Nn) / 128, 256, 0, stream>>>(Rx, S1, S2, Wcbf, bc,
                                                             wpre, wadp, Cg);
  // --- combine ---
  final_kernel<<<(int)((BND + 255) / 256), 256, 0, stream>>>(Zg, hidden, Cg, out, (int)BND);
}
